// A3TGCN_14267881357858
// MI455X (gfx1250) — matrix operand order repaired and checked by reading, not compiled, not run on hardware
//
#include <hip/hip_runtime.h>
#include <hip/hip_bf16.h>
#include <math.h>

// Problem constants (from reference)
constexpr int N_  = 10000;
constexpr int E_  = 320000;
constexpr int F_  = 128;
constexpr int H_  = 128;
constexpr int T_  = 16;
constexpr int L_  = 2;
constexpr int O_  = 128;
constexpr int H3_ = 3 * H_;   // fused z|r|h conv columns
constexpr int KK_ = 128;      // every GEMM here has K == 128

typedef __attribute__((ext_vector_type(16))) _Float16 v16h;
typedef __attribute__((ext_vector_type(8)))  _Float16 v8h;
typedef __attribute__((ext_vector_type(8)))  float    v8f;

// ---------------------------------------------------------------------------
// f16 WMMA GEMM, K fixed at 128, B pre-swizzled into fragment layout.
//   C(M x Ncol) = A(M x K) * B(K x Ncol) [+ bias] [+ C]
// A row-major f16 (lda); Bfrag: per (ktile, ntile) 16x32 tile stored as
// 32 lanes x 16 halves contiguously, ktile-major then ntile:
//   frag[((kk*(Ncol/16) + tn)*32 + lane)] as v16h, where element j of lane is
//   B[(kk*32 + (lane>>4)*16 + j)*Ncol + tn*16 + (lane&15)]
// One wave computes a 16x64 C tile (4 n-tiles), K unrolled (4 WMMA steps x 4).
// ---------------------------------------------------------------------------
__global__ __launch_bounds__(256) void k_gemm_f16(
    const _Float16* __restrict__ A, int lda,
    const _Float16* __restrict__ Bfrag,
    float* __restrict__ C, int ldc,
    const float* __restrict__ bias,
    int M, int Ncol, int acc)
{
  constexpr int NT = 4;                       // n-tiles per wave
  const int wave = (blockIdx.x * blockDim.x + threadIdx.x) >> 5;
  const int lane = threadIdx.x & 31;
  const int ngroups = Ncol / (16 * NT);
  const int total   = (M >> 4) * ngroups;
  if (wave >= total) return;                  // wave-uniform: EXEC all-ones
  const int tm = wave / ngroups;
  const int tg = wave % ngroups;
  const int l15 = lane & 15;
  const int hi  = lane >> 4;
  const int ntiles = Ncol >> 4;

  const _Float16* Arow = A + (size_t)((tm << 4) + l15) * lda;
  const v16h* Bf = (const v16h*)Bfrag + (size_t)tg * NT * 32 + lane;

  v8f c0 = {0.f,0.f,0.f,0.f,0.f,0.f,0.f,0.f};
  v8f c1 = c0, c2 = c0, c3 = c0;

#pragma unroll
  for (int kk = 0; kk < KK_ / 32; ++kk) {
    const int kb = kk * 32;
    // A 16x32 fragment: lane<16 -> K {kb..kb+15}
    //                   lane>=16 -> K {kb+16..kb+31}
    v16h a  = *(const v16h*)(Arow + kb + hi * 16);
    const v16h* Bk = Bf + (size_t)kk * ntiles * 32;
    v16h b0 = Bk[0 * 32];
    v16h b1 = Bk[1 * 32];
    v16h b2 = Bk[2 * 32];
    v16h b3 = Bk[3 * 32];
    c0 = __builtin_amdgcn_wmma_f32_16x16x32_f16(false, a, false, b0,
                                                (short)0, c0, false, false);
    c1 = __builtin_amdgcn_wmma_f32_16x16x32_f16(false, a, false, b1,
                                                (short)0, c1, false, false);
    c2 = __builtin_amdgcn_wmma_f32_16x16x32_f16(false, a, false, b2,
                                                (short)0, c2, false, false);
    c3 = __builtin_amdgcn_wmma_f32_16x16x32_f16(false, a, false, b3,
                                                (short)0, c3, false, false);
  }

  // C layout per tile: VGPR r -> M = tm*16 + hi*8 + r, N = tile_col + l15
  const int m0 = (tm << 4) + hi * 8;
  v8f acc_[4] = {c0, c1, c2, c3};
#pragma unroll
  for (int u = 0; u < NT; ++u) {
    const int col = ((tg * NT + u) << 4) + l15;
    const float badd = bias ? bias[col] : 0.0f;
#pragma unroll
    for (int r = 0; r < 8; ++r) {
      size_t o = (size_t)(m0 + r) * ldc + col;
      float v = acc_[u][r] + badd;
      if (acc) v += C[o];
      C[o] = v;
    }
  }
}

// B fragment pre-swizzle: row-major f16 (K x Ncol) -> fragment layout above
__global__ void k_fragB(const _Float16* __restrict__ src,
                        _Float16* __restrict__ dst, int K, int Ncol) {
  int i = blockIdx.x * blockDim.x + threadIdx.x;
  if (i >= K * Ncol) return;
  int tile = i >> 9;            // /512
  int r    = i & 511;
  int lane = r >> 4;
  int j    = r & 15;
  int ntiles = Ncol >> 4;
  int kk = tile / ntiles;
  int tn = tile % ntiles;
  int krow = kk * 32 + (lane >> 4) * 16 + j;
  int col  = tn * 16 + (lane & 15);
  dst[i] = src[(size_t)krow * Ncol + col];
}

// ---------------------------- setup / elementwise ---------------------------
__global__ void k_zero(float* p, int n) {
  int i = blockIdx.x * blockDim.x + threadIdx.x; if (i < n) p[i] = 0.0f;
}
__global__ void k_fill(float* p, float v, int n) {
  int i = blockIdx.x * blockDim.x + threadIdx.x; if (i < n) p[i] = v;
}
__global__ void k_deg(const int* __restrict__ dst, const float* __restrict__ ew,
                      float* deg, int e) {
  int i = blockIdx.x * blockDim.x + threadIdx.x;
  if (i < e) atomicAdd(&deg[dst[i]], ew[i]);
}
__global__ void k_dinv(float* d, int n) {
  int i = blockIdx.x * blockDim.x + threadIdx.x;
  if (i < n) d[i] = rsqrtf(d[i] + 1.0f);
}
__global__ void k_norme(const int* __restrict__ src, const int* __restrict__ dst,
                        const float* __restrict__ ew, const float* __restrict__ dinv,
                        float* ne, int e) {
  int i = blockIdx.x * blockDim.x + threadIdx.x;
  if (i < e) ne[i] = dinv[src[i]] * ew[i] * dinv[dst[i]];
}
__global__ void k_cvt(const float* __restrict__ s, _Float16* __restrict__ d, int n) {
  int i = blockIdx.x * blockDim.x + threadIdx.x;
  if (i < n) d[i] = (_Float16)s[i];
}
// x is (N,F,T); x_t[n,f] = x[(n*F+f)*T + t]
__global__ void k_cvt_x(const float* __restrict__ x, _Float16* __restrict__ d,
                        int t, int n) {
  int i = blockIdx.x * blockDim.x + threadIdx.x;
  if (i < n) d[i] = (_Float16)x[(size_t)i * T_ + t];
}
// Pack [Wcz | Wcr | Wch] -> (L, H, 3H) row-major f16
__global__ void k_pack_wc3(const float* __restrict__ Wz, const float* __restrict__ Wr,
                           const float* __restrict__ Wh, _Float16* __restrict__ out) {
  int i = blockIdx.x * blockDim.x + threadIdx.x;
  if (i >= L_ * H_ * H3_) return;
  int l = i / (H_ * H3_); int rem = i % (H_ * H3_);
  int row = rem / H3_;    int colc = rem % H3_;
  int g = colc / H_;      int c = colc % H_;
  const float* W = (g == 0) ? Wz : (g == 1) ? Wr : Wh;
  out[i] = (_Float16)W[(size_t)l * H_ * H_ + (size_t)row * H_ + c];
}
// Split Wl (L,2H,H) into top (rows 0..H-1) and bottom (rows H..2H-1), 3 gates
__global__ void k_pack_wl(const float* __restrict__ Wz, const float* __restrict__ Wr,
                          const float* __restrict__ Wh,
                          _Float16* __restrict__ top, _Float16* __restrict__ bot) {
  int i = blockIdx.x * blockDim.x + threadIdx.x;
  if (i >= 3 * L_ * H_ * H_) return;
  int g = i / (L_ * H_ * H_); int rem = i % (L_ * H_ * H_);
  int l = rem / (H_ * H_);    int rr = rem % (H_ * H_);
  int row = rr / H_;          int c = rr % H_;
  const float* W = (g == 0) ? Wz : (g == 1) ? Wr : Wh;
  size_t base = (size_t)l * 2 * H_ * H_;
  top[i] = (_Float16)W[base + (size_t)row * H_ + c];
  bot[i] = (_Float16)W[base + (size_t)(row + H_) * H_ + c];
}
__global__ void k_pack_b3(const float* __restrict__ bz, const float* __restrict__ br,
                          const float* __restrict__ bh, float* __restrict__ out) {
  int i = blockIdx.x * blockDim.x + threadIdx.x;
  if (i >= L_ * H3_) return;
  int l = i / H3_; int c = i % H3_;
  int g = c / H_;  int k = c % H_;
  const float* b = (g == 0) ? bz : (g == 1) ? br : bh;
  out[i] = b[l * H_ + k];
}
// Edge scatter: agg[dst,:] += norm_e * hw[src,:], 4 columns per thread
__global__ void k_scatter(const int* __restrict__ src, const int* __restrict__ dst,
                          const float* __restrict__ ne, const float* __restrict__ hw,
                          float* __restrict__ agg) {
  int idx = blockIdx.x * blockDim.x + threadIdx.x;
  if (idx >= E_ * (H3_ / 4)) return;
  int e  = idx / (H3_ / 4);
  int cg = (idx % (H3_ / 4)) * 4;
  float w = ne[e];
  size_t sb = (size_t)src[e] * H3_ + cg;
  size_t db = (size_t)dst[e] * H3_ + cg;
  __builtin_prefetch(&hw[sb], 0, 1);   // global_prefetch_b8 on gathered row
#pragma unroll
  for (int j = 0; j < 4; ++j) atomicAdd(&agg[db + j], w * hw[sb + j]);
}
// conv = agg + dinv^2 * hw + bias  -> f16
__global__ void k_conv3(const float* __restrict__ agg, const float* __restrict__ hw,
                        const float* __restrict__ dinv, const float* __restrict__ b3,
                        _Float16* __restrict__ out) {
  int i = blockIdx.x * blockDim.x + threadIdx.x;
  if (i >= N_ * H3_) return;
  int n = i / H3_; int c = i % H3_;
  float sn = dinv[n];
  out[i] = (_Float16)(agg[i] + sn * sn * hw[i] + b3[c]);
}
// z = sigmoid(pre_z) in place; rhp = sigmoid(pre_r) * hp  -> f16
__global__ void k_sig_rhp(float* __restrict__ pz, const float* __restrict__ pr,
                          const float* __restrict__ hp, _Float16* __restrict__ rhp,
                          int n) {
  int i = blockIdx.x * blockDim.x + threadIdx.x;
  if (i >= n) return;
  float z = 1.0f / (1.0f + __expf(-pz[i]));
  pz[i] = z;
  float r = 1.0f / (1.0f + __expf(-pr[i]));
  rhp[i] = (_Float16)(r * hp[i]);
}
// h_new = z*hp + (1-z)*tanh(pre_h); write f32 state + f16 copy for next layer
__global__ void k_gru(const float* __restrict__ z, const float* __restrict__ preh,
                      float* __restrict__ hprev, _Float16* __restrict__ hin, int n) {
  int i = blockIdx.x * blockDim.x + threadIdx.x;
  if (i >= n) return;
  float ht = tanhf(preh[i]);
  float zz = z[i];
  float hn = zz * hprev[i] + (1.0f - zz) * ht;
  hprev[i] = hn;
  hin[i] = (_Float16)hn;
}
// Online-softmax attention update, one wave per node
__global__ __launch_bounds__(256) void k_attn(
    const float* __restrict__ h, const float* __restrict__ aW,
    const float* __restrict__ ab, float* __restrict__ m,
    float* __restrict__ den, float* __restrict__ ctx) {
  int wid  = (blockIdx.x * blockDim.x + threadIdx.x) >> 5;
  int lane = threadIdx.x & 31;
  if (wid >= N_) return;
  const float* hr = h + (size_t)wid * H_;
  float part = 0.0f;
  for (int i = lane; i < H_; i += 32) part += hr[i] * aW[i];
#pragma unroll
  for (int off = 16; off > 0; off >>= 1) part += __shfl_xor(part, off, 32);
  float s  = part + ab[0];
  float mo = m[wid];
  float mn = fmaxf(mo, s);
  float alpha = __expf(mo - mn);   // exp(-inf) = 0 on first step
  float beta  = __expf(s - mn);
  if (lane == 0) { m[wid] = mn; den[wid] = den[wid] * alpha + beta; }
  for (int i = lane; i < H_; i += 32) {
    size_t o = (size_t)wid * H_ + i;
    ctx[o] = ctx[o] * alpha + beta * hr[i];
  }
}
__global__ void k_ctxnorm(const float* __restrict__ ctx, const float* __restrict__ den,
                          _Float16* __restrict__ out, int n) {
  int i = blockIdx.x * blockDim.x + threadIdx.x;
  if (i >= n) return;
  out[i] = (_Float16)(ctx[i] / den[i / H_]);
}

// ---------------------------------------------------------------------------
extern "C" void kernel_launch(void* const* d_in, const int* in_sizes, int n_in,
                              void* d_out, int out_size, void* d_ws, size_t ws_size,
                              hipStream_t stream) {
  (void)in_sizes; (void)n_in; (void)out_size; (void)ws_size;

  const float* x    = (const float*)d_in[0];
  const int*   ei   = (const int*)d_in[1];
  const float* ew   = (const float*)d_in[2];
  const float* Wcz  = (const float*)d_in[3];
  const float* bcz  = (const float*)d_in[4];
  const float* Wlz  = (const float*)d_in[5];
  const float* blz  = (const float*)d_in[6];
  const float* Wcr  = (const float*)d_in[7];
  const float* bcr  = (const float*)d_in[8];
  const float* Wlr  = (const float*)d_in[9];
  const float* blr  = (const float*)d_in[10];
  const float* Wch  = (const float*)d_in[11];
  const float* bch  = (const float*)d_in[12];
  const float* Wlh  = (const float*)d_in[13];
  const float* blh  = (const float*)d_in[14];
  const float* attW = (const float*)d_in[15];
  const float* attb = (const float*)d_in[16];
  const float* outW = (const float*)d_in[17];
  const float* outb = (const float*)d_in[18];
  const int* src = ei;
  const int* dst = ei + E_;

  // Workspace carve-up (256B aligned)
  char* ws = (char*)d_ws;
  size_t off = 0;
  auto carve = [&](size_t bytes) -> char* {
    size_t o = (off + 255) & ~(size_t)255;
    off = o + bytes;
    return ws + o;
  };
  float*    dinv    = (float*)carve((size_t)N_ * 4);
  float*    norme   = (float*)carve((size_t)E_ * 4);
  _Float16* Wc3h    = (_Float16*)carve((size_t)L_ * H_ * H3_ * 2);
  _Float16* WlT     = (_Float16*)carve((size_t)3 * L_ * H_ * H_ * 2);
  _Float16* WlB     = (_Float16*)carve((size_t)3 * L_ * H_ * H_ * 2);
  float*    bias3   = (float*)carve((size_t)L_ * H3_ * 4);
  _Float16* outWh   = (_Float16*)carve((size_t)H_ * O_ * 2);
  _Float16* Wc3f    = (_Float16*)carve((size_t)L_ * H_ * H3_ * 2);   // fragments
  _Float16* WlTf    = (_Float16*)carve((size_t)3 * L_ * H_ * H_ * 2);
  _Float16* WlBf    = (_Float16*)carve((size_t)3 * L_ * H_ * H_ * 2);
  _Float16* outWf   = (_Float16*)carve((size_t)H_ * O_ * 2);
  float*    hprev   = (float*)carve((size_t)L_ * N_ * H_ * 4);
  _Float16* hin_h   = (_Float16*)carve((size_t)N_ * H_ * 2);
  _Float16* hp_h    = (_Float16*)carve((size_t)N_ * H_ * 2);
  float*    hw3     = (float*)carve((size_t)N_ * H3_ * 4);
  float*    agg3    = (float*)carve((size_t)N_ * H3_ * 4);
  _Float16* conv3h  = (_Float16*)carve((size_t)N_ * H3_ * 2);
  float*    attM    = (float*)carve((size_t)N_ * 4);
  float*    attD    = (float*)carve((size_t)N_ * 4);
  float*    ctx     = (float*)carve((size_t)N_ * H_ * 4);
  _Float16* ctx_h   = (_Float16*)carve((size_t)N_ * H_ * 2);
  // Aliases of buffers that are dead by the time these are written:
  float*    pre_z   = hw3;                 // N*H each, 3 fit exactly in hw3
  float*    pre_r   = hw3 + (size_t)N_ * H_;
  float*    pre_h   = hw3 + (size_t)2 * N_ * H_;
  _Float16* rhp_h   = (_Float16*)agg3;     // agg3 dead after k_conv3

  const int TPB = 256;
  auto grid = [&](long n) { return dim3((unsigned)((n + TPB - 1) / TPB)); };
  auto gemm = [&](const _Float16* A, int lda, const _Float16* Bfrag,
                  float* C, int ldc, const float* bias,
                  int M, int Ncol, int acc) {
    long threads = (long)(M / 16) * (Ncol / 64) * 32;
    k_gemm_f16<<<grid(threads), TPB, 0, stream>>>(A, lda, Bfrag, C, ldc,
                                                  bias, M, Ncol, acc);
  };
  auto fragB = [&](const _Float16* srcB, _Float16* dstB, int K, int Ncol) {
    k_fragB<<<grid((long)K * Ncol), TPB, 0, stream>>>(srcB, dstB, K, Ncol);
  };

  // ---- graph normalization ----
  k_zero<<<grid(N_), TPB, 0, stream>>>(dinv, N_);
  k_deg<<<grid(E_), TPB, 0, stream>>>(dst, ew, dinv, E_);
  k_dinv<<<grid(N_), TPB, 0, stream>>>(dinv, N_);
  k_norme<<<grid(E_), TPB, 0, stream>>>(src, dst, ew, dinv, norme, E_);

  // ---- weight packing (f16 row-major, then WMMA-fragment swizzle) ----
  k_pack_wc3<<<grid((long)L_ * H_ * H3_), TPB, 0, stream>>>(Wcz, Wcr, Wch, Wc3h);
  k_pack_wl<<<grid((long)3 * L_ * H_ * H_), TPB, 0, stream>>>(Wlz, Wlr, Wlh, WlT, WlB);
  k_pack_b3<<<grid((long)L_ * H3_), TPB, 0, stream>>>(bcz, bcr, bch, bias3);
  k_cvt<<<grid((long)H_ * O_), TPB, 0, stream>>>(outW, outWh, H_ * O_);
  const size_t gsz = (size_t)H_ * H_;
  for (int l = 0; l < L_; ++l)
    fragB(Wc3h + (size_t)l * H_ * H3_, Wc3f + (size_t)l * H_ * H3_, H_, H3_);
  for (int gl = 0; gl < 3 * L_; ++gl) {
    fragB(WlT + gl * gsz, WlTf + gl * gsz, H_, H_);
    fragB(WlB + gl * gsz, WlBf + gl * gsz, H_, H_);
  }
  fragB(outWh, outWf, H_, O_);

  // ---- state init (every call: deterministic) ----
  k_zero<<<grid((long)L_ * N_ * H_), TPB, 0, stream>>>(hprev, L_ * N_ * H_);
  k_zero<<<grid((long)N_ * H_), TPB, 0, stream>>>(ctx, N_ * H_);
  k_zero<<<grid(N_), TPB, 0, stream>>>(attD, N_);
  k_fill<<<grid(N_), TPB, 0, stream>>>(attM, -INFINITY, N_);

  // ---- time loop ----
  for (int t = 0; t < T_; ++t) {
    for (int l = 0; l < L_; ++l) {
      float* hp = hprev + (size_t)l * N_ * H_;
      if (l == 0)
        k_cvt_x<<<grid((long)N_ * F_), TPB, 0, stream>>>(x, hin_h, t, N_ * F_);
      // else hin_h already holds layer l-1's new state (written by k_gru)
      k_cvt<<<grid((long)N_ * H_), TPB, 0, stream>>>(hp, hp_h, N_ * H_);

      // fused z|r|h conv GEMM: hw3 = h_in @ [Wcz Wcr Wch]
      gemm(hin_h, H_, Wc3f + (size_t)l * H_ * H3_, hw3, H3_,
           nullptr, N_, H3_, 0);

      // sparse aggregation
      k_zero<<<grid((long)N_ * H3_), TPB, 0, stream>>>(agg3, N_ * H3_);
      k_scatter<<<grid((long)E_ * (H3_ / 4)), TPB, 0, stream>>>(src, dst, norme,
                                                                hw3, agg3);
      k_conv3<<<grid((long)N_ * H3_), TPB, 0, stream>>>(agg3, hw3, dinv,
                                                        bias3 + l * H3_, conv3h);

      // z / r gates: conv_g @ Wl_top + hp @ Wl_bot
      gemm(conv3h + 0 * H_, H3_, WlTf + ((size_t)0 * L_ + l) * gsz,
           pre_z, H_, blz + l * H_, N_, H_, 0);
      gemm(hp_h, H_, WlBf + ((size_t)0 * L_ + l) * gsz,
           pre_z, H_, nullptr, N_, H_, 1);
      gemm(conv3h + 1 * H_, H3_, WlTf + ((size_t)1 * L_ + l) * gsz,
           pre_r, H_, blr + l * H_, N_, H_, 0);
      gemm(hp_h, H_, WlBf + ((size_t)1 * L_ + l) * gsz,
           pre_r, H_, nullptr, N_, H_, 1);

      k_sig_rhp<<<grid((long)N_ * H_), TPB, 0, stream>>>(pre_z, pre_r, hp,
                                                         rhp_h, N_ * H_);

      // h gate: conv_h @ Wl_top + (r*hp) @ Wl_bot
      gemm(conv3h + 2 * H_, H3_, WlTf + ((size_t)2 * L_ + l) * gsz,
           pre_h, H_, blh + l * H_, N_, H_, 0);
      gemm(rhp_h, H_, WlBf + ((size_t)2 * L_ + l) * gsz,
           pre_h, H_, nullptr, N_, H_, 1);

      k_gru<<<grid((long)N_ * H_), TPB, 0, stream>>>(pre_z, pre_h, hp, hin_h,
                                                     N_ * H_);
      if (l == L_ - 1)
        k_attn<<<grid((long)N_ * 32), TPB, 0, stream>>>(hp, attW, attb,
                                                        attM, attD, ctx);
    }
  }

  // ---- attention context -> output projection ----
  k_ctxnorm<<<grid((long)N_ * H_), TPB, 0, stream>>>(ctx, attD, ctx_h, N_ * H_);
  gemm(ctx_h, H_, outWf, (float*)d_out, O_, outb, N_, O_, 0);
}
